// ScaledDotProductAttention_8778913153179
// MI455X (gfx1250) — matrix operand order repaired and checked by reading, not compiled, not run on hardware
//
#include <hip/hip_runtime.h>

// ---------------------------------------------------------------------------
// Types for CDNA5 WMMA (wave32): v_wmma_f32_16x16x32_bf16
// ---------------------------------------------------------------------------
typedef __attribute__((ext_vector_type(16))) __bf16 v16bf;
typedef __attribute__((ext_vector_type(8)))  __bf16 v8bf;
typedef __attribute__((ext_vector_type(4)))  __bf16 v4bf;
typedef __attribute__((ext_vector_type(8)))  float  v8f;
typedef __attribute__((ext_vector_type(4)))  float  v4f;

#define WMMA_BF16(A, B, C) \
  __builtin_amdgcn_wmma_f32_16x16x32_bf16(false, (A), false, (B), (short)0, (C), false, false)

// Problem constants (from reference): B=32, Lp=Lq=1024, Lv=D=512
#define NB   32
#define LQ   1024
#define DD   512
#define PLANE 16777216u              // 32*1024*512 elements per output tensor
#define SCALE 0.044194173824159216f // 1/sqrt(512)

// ---------------------------------------------------------------------------
// Async global->LDS staging of one contiguous block (CDNA5 async-copy path,
// tracked by ASYNCcnt).  256 threads, 16B per thread per wavefront of copies.
// ---------------------------------------------------------------------------
__device__ __forceinline__ void stage_tile_async(const __bf16* g, __bf16* l,
                                                 int tid, int bytes) {
  unsigned lds_off = (unsigned)(uintptr_t)l + (unsigned)tid * 16u;
  const char* gp = (const char*)g + (size_t)tid * 16u;
  for (int off = 0; off < bytes; off += 4096) {
    asm volatile("global_load_async_to_lds_b128 %0, %1, off"
                 :
                 : "v"(lds_off + (unsigned)off), "v"(gp + off)
                 : "memory");
  }
  asm volatile("s_wait_asynccnt 0x0" ::: "memory");
}

// ---------------------------------------------------------------------------
// A-fragment: 16x32 bf16 from an f32 row-major tile (LDS scores), cvt on the
// fly.  Lane m=L&15; lanes 0-15 hold K 0-7/16-23, lanes 16-31 hold
// K 8-15/24-31.  p0 = tile_base + k0.
// ---------------------------------------------------------------------------
__device__ __forceinline__ v16bf load_a_f32(const float* p0, int lane, int ld) {
  const float* p = p0 + (size_t)(lane & 15) * ld + ((lane >> 4) << 3);
  const v4f* pv = (const v4f*)p;
  v4f x0 = pv[0], x1 = pv[1], x2 = pv[4], x3 = pv[5];
  v16bf r;
#pragma unroll
  for (int i = 0; i < 4; ++i) {
    r[i]      = (__bf16)x0[i];
    r[i + 4]  = (__bf16)x1[i];
    r[i + 8]  = (__bf16)x2[i];
    r[i + 12] = (__bf16)x3[i];
  }
  return r;
}

// A-fragment from a bf16 row-major tile (LDS): 2x ds_load_b128.
__device__ __forceinline__ v16bf load_a_bf16(const __bf16* p0, int lane, int ld) {
  const __bf16* p = p0 + (size_t)(lane & 15) * ld + ((lane >> 4) << 3);
  v8bf lo = *(const v8bf*)(p);       // K 0-7  (or 8-15)
  v8bf hi = *(const v8bf*)(p + 16);  // K 16-23 (or 24-31)
  v16bf r;
#pragma unroll
  for (int i = 0; i < 8; ++i) { r[i] = lo[i]; r[i + 8] = hi[i]; }
  return r;
}

// B-fragment: 32x16 bf16 from a "B-transposed" row-major bf16 matrix (row n
// holds column n of B over K).  Lane n=L&15; lanes 0-15 hold K 0-7/16-23,
// lanes 16-31 hold K 8-15/24-31.  p0 = base + n0*ld + k0.
__device__ __forceinline__ v16bf load_b_bf16(const __bf16* p0, int lane, int ld) {
  const __bf16* p = p0 + (size_t)(lane & 15) * ld + ((lane >> 4) << 3);
  v8bf lo = *(const v8bf*)(p);
  v8bf hi = *(const v8bf*)(p + 16);
  v16bf r;
#pragma unroll
  for (int i = 0; i < 8; ++i) { r[i] = lo[i]; r[i + 8] = hi[i]; }
  return r;
}

// ---------------------------------------------------------------------------
// Convert f32 [B, R, 512] -> bf16 row-major [B, R, 512] AND bf16 transposed
// [B, 512, R].  32x32 LDS tile, fully coalesced both directions.
// ---------------------------------------------------------------------------
__global__ void __launch_bounds__(256)
cvt_tr_kernel(const float* __restrict__ in, __bf16* __restrict__ rm,
              __bf16* __restrict__ tr, int R) {
  __shared__ __bf16 tile[32][33];
  const int b  = blockIdx.z;
  const int r0 = blockIdx.x * 32;
  const int c0 = blockIdx.y * 32;
  const int c  = threadIdx.x & 31;
  const int rr = threadIdx.x >> 5;
  const float*  src = in + ((size_t)b * R + r0) * DD + c0;
  __bf16*       rmo = rm + ((size_t)b * R + r0) * DD + c0;
#pragma unroll
  for (int i = 0; i < 4; ++i) {
    int r = rr + i * 8;
    __bf16 h = (__bf16)src[(size_t)r * DD + c];
    tile[r][c] = h;
    rmo[(size_t)r * DD + c] = h;
  }
  __syncthreads();
  __bf16* tro = tr + ((size_t)b * DD + c0) * R + r0;
#pragma unroll
  for (int i = 0; i < 4; ++i) {
    int r = rr + i * 8;                       // transposed row = original col
    tro[(size_t)r * R + c] = tile[c][r];
  }
}

// Plain f32 -> bf16 transcode (for `pre`), float4 in / bf16x4 out.
__global__ void __launch_bounds__(256)
cvt_kernel(const float* __restrict__ in, __bf16* __restrict__ outb) {
  const size_t i = (size_t)blockIdx.x * 256 + threadIdx.x;
  v4f x = ((const v4f*)in)[i];
  v4bf y;
#pragma unroll
  for (int k = 0; k < 4; ++k) y[k] = (__bf16)x[k];
  ((v4bf*)outb)[i] = y;
}

// ---------------------------------------------------------------------------
// Vectorized in-place row softmax over LDS scores.  Thread (r = tid>>4,
// c = tid&15) owns float4 columns [c + 16j]; width-16 shuffle reductions;
// exp() left unnormalized, 1/sum stored per row.
// ---------------------------------------------------------------------------
__device__ __forceinline__ void softmax_rows(float* Sbuf, float* inv_sum,
                                             int tid, int W) {
  const int r = tid >> 4, c = tid & 15;
  v4f* row = (v4f*)(Sbuf + (size_t)r * W);
  const int nv = W >> 2;  // float4s per row
  float mx = -3.0e38f;
  for (int j = c; j < nv; j += 16) {
    v4f x = row[j];
    mx = fmaxf(mx, fmaxf(fmaxf(x[0], x[1]), fmaxf(x[2], x[3])));
  }
#pragma unroll
  for (int o = 8; o >= 1; o >>= 1) mx = fmaxf(mx, __shfl_xor(mx, o, 16));
  float s = 0.f;
  for (int j = c; j < nv; j += 16) {
    v4f x = row[j];
    v4f e;
#pragma unroll
    for (int i = 0; i < 4; ++i) { e[i] = __expf(x[i] - mx); s += e[i]; }
    row[j] = e;
  }
#pragma unroll
  for (int o = 8; o >= 1; o >>= 1) s += __shfl_xor(s, o, 16);
  if (c == 0) inv_sum[r] = 1.f / s;
}

// ---------------------------------------------------------------------------
// Stream 1: precompute = softmax(pre.q^T * s) . q
// One 16-row tile of Lp per block (256 thr = 8 waves).  LDS: 16x1024 f32
// scores + 16x512 bf16 A tile (async-staged) + row inv-sums.
// ---------------------------------------------------------------------------
__global__ void __launch_bounds__(256)
attn_pre_kernel(const __bf16* __restrict__ preb, const __bf16* __restrict__ qb,
                const __bf16* __restrict__ qTb, float* __restrict__ outPC) {
  extern __shared__ __align__(16) char smem[];
  float*  Sbuf    = (float*)smem;                     // 16 x 1024 f32 (64 KB)
  __bf16* Abuf    = (__bf16*)(smem + 65536);          // 16 x 512 bf16 (16 KB)
  float*  inv_sum = (float*)(smem + 65536 + 16384);   // 16 f32
  const int tid = threadIdx.x, lane = tid & 31, wave = tid >> 5;
  const int batch = blockIdx.y, row0 = blockIdx.x * 16;
  const __bf16* qB  = qb  + (size_t)batch * LQ * DD;  // row-major bf16 q
  const __bf16* qTB = qTb + (size_t)batch * DD * LQ;  // transposed bf16 q

  // ---- phase 0: async-DMA the 16x512 bf16 A tile (contiguous 16 KB) ----
  stage_tile_async(preb + ((size_t)batch * LQ + row0) * DD, Abuf, tid, 16384);
  __syncthreads();

  // ---- phase 1: S = scale * (pre . q^T); wave owns 128 cols of Lq ----
  // One-tile-lookahead rotation on B so each WMMA runs with the next B pair
  // already in flight (avoids full s_wait_loadcnt 0x0 per tile).
  {
    v8f acc[8];
#pragma unroll
    for (int t = 0; t < 8; ++t) acc[t] = {};
    const int n0 = wave * 128;
    v16bf bc = load_b_bf16(qB + (size_t)n0 * DD, lane, DD);
    for (int kt = 0; kt < 16; ++kt) {
      const int k0 = kt * 32;
      v16bf a = load_a_bf16(Abuf + k0, lane, DD);
#pragma unroll
      for (int t = 0; t < 8; ++t) {
        const int tn = (t + 1) & 7;
        const int kn = (t == 7) ? ((kt + 1 < 16) ? (kt + 1) * 32 : k0) : k0;
        v16bf bn = load_b_bf16(qB + (size_t)(n0 + tn * 16) * DD + kn, lane, DD);
        acc[t] = WMMA_BF16(a, bc, acc[t]);
        bc = bn;
      }
    }
    const int n = lane & 15, mb = (lane >> 4) << 3;
#pragma unroll
    for (int t = 0; t < 8; ++t)
#pragma unroll
      for (int i = 0; i < 8; ++i)
        Sbuf[(mb + i) * 1024 + n0 + t * 16 + n] = acc[t][i] * SCALE;
  }
  __syncthreads();

  // ---- phase 2: row softmax (exp kept unnormalized in place) ----
  softmax_rows(Sbuf, inv_sum, tid, 1024);
  __syncthreads();

  // ---- phase 3: PC = P . q (K = Lq = 1024); wave owns 64 cols of D ----
  // Software-pipelined one K-step ahead (unroll-2 renames the rotation).
  {
    v8f acc[4];
#pragma unroll
    for (int t = 0; t < 4; ++t) acc[t] = {};
    const int n0 = wave * 64;
    v16bf a0 = load_a_f32(Sbuf, lane, 1024);
    v16bf b0[4];
#pragma unroll
    for (int t = 0; t < 4; ++t)
      b0[t] = load_b_bf16(qTB + (size_t)(n0 + t * 16) * LQ, lane, LQ);
#pragma unroll 2
    for (int kt = 0; kt < 32; ++kt) {
      const int kn = (kt + 1 < 32) ? (kt + 1) * 32 : kt * 32;
      v16bf a1 = load_a_f32(Sbuf + kn, lane, 1024);
      v16bf b1[4];
#pragma unroll
      for (int t = 0; t < 4; ++t)
        b1[t] = load_b_bf16(qTB + (size_t)(n0 + t * 16) * LQ + kn, lane, LQ);
#pragma unroll
      for (int t = 0; t < 4; ++t)
        acc[t] = WMMA_BF16(a0, b0[t], acc[t]);
      a0 = a1;
#pragma unroll
      for (int t = 0; t < 4; ++t) b0[t] = b1[t];
    }
    const int n = lane & 15, mb = (lane >> 4) << 3;
#pragma unroll
    for (int t = 0; t < 4; ++t)
#pragma unroll
      for (int i = 0; i < 8; ++i) {
        const int m = mb + i;
        outPC[((size_t)batch * LQ + row0 + m) * DD + n0 + t * 16 + n] =
            acc[t][i] * inv_sum[m];
      }
  }
}

// ---------------------------------------------------------------------------
// Stream 2: attne = softmax(q.v^T * s); energy = attne.v; wc = energy.v
// One 16-row tile of Lq per block.  LDS: 16x512 f32 scores + 16x512 bf16
// energy staging + 16x512 bf16 A tile (async-staged).
// ---------------------------------------------------------------------------
__global__ void __launch_bounds__(256)
attn_ev_kernel(const __bf16* __restrict__ qbrm, const __bf16* __restrict__ vb,
               const __bf16* __restrict__ vTb, float* __restrict__ outWC,
               float* __restrict__ outEN) {
  extern __shared__ __align__(16) char smem[];
  float*  Sbuf    = (float*)smem;                        // 16 x 512 f32
  __bf16* ebuf    = (__bf16*)(smem + 32768);             // 16 x 512 bf16
  __bf16* Abuf    = (__bf16*)(smem + 32768 + 16384);     // 16 x 512 bf16
  float*  inv_sum = (float*)(smem + 32768 + 2 * 16384);  // 16 f32
  const int tid = threadIdx.x, lane = tid & 31, wave = tid >> 5;
  const int batch = blockIdx.y, row0 = blockIdx.x * 16;
  const __bf16* vB  = vb  + (size_t)batch * DD * DD;   // row-major bf16 v
  const __bf16* vTB = vTb + (size_t)batch * DD * DD;   // transposed bf16 v
  const int n0 = wave * 64;
  const int n = lane & 15, mb = (lane >> 4) << 3;

  // ---- phase 0: async-DMA the 16x512 bf16 q tile (contiguous 16 KB) ----
  stage_tile_async(qbrm + ((size_t)batch * LQ + row0) * DD, Abuf, tid, 16384);
  __syncthreads();

  // ---- phase 1: S = scale * (q . v^T); wave owns 64 cols of Lv ----
  {
    v8f acc[4];
#pragma unroll
    for (int t = 0; t < 4; ++t) acc[t] = {};
    v16bf a0 = load_a_bf16(Abuf, lane, DD);
    v16bf b0[4];
#pragma unroll
    for (int t = 0; t < 4; ++t)
      b0[t] = load_b_bf16(vB + (size_t)(n0 + t * 16) * DD, lane, DD);
#pragma unroll 2
    for (int kt = 0; kt < 16; ++kt) {
      const int kn = (kt + 1 < 16) ? (kt + 1) * 32 : kt * 32;
      v16bf a1 = load_a_bf16(Abuf + kn, lane, DD);
      v16bf b1[4];
#pragma unroll
      for (int t = 0; t < 4; ++t)
        b1[t] = load_b_bf16(vB + (size_t)(n0 + t * 16) * DD + kn, lane, DD);
#pragma unroll
      for (int t = 0; t < 4; ++t)
        acc[t] = WMMA_BF16(a0, b0[t], acc[t]);
      a0 = a1;
#pragma unroll
      for (int t = 0; t < 4; ++t) b0[t] = b1[t];
    }
#pragma unroll
    for (int t = 0; t < 4; ++t)
#pragma unroll
      for (int i = 0; i < 8; ++i)
        Sbuf[(mb + i) * 512 + n0 + t * 16 + n] = acc[t][i] * SCALE;
  }
  __syncthreads();

  // ---- phase 2: row softmax over 512 ----
  softmax_rows(Sbuf, inv_sum, tid, 512);
  __syncthreads();

  // ---- phase 3: EN = attne . v  -> HBM (f32) and LDS (bf16) ----
  {
    v8f acc[4];
#pragma unroll
    for (int t = 0; t < 4; ++t) acc[t] = {};
    v16bf a0 = load_a_f32(Sbuf, lane, 512);
    v16bf b0[4];
#pragma unroll
    for (int t = 0; t < 4; ++t)
      b0[t] = load_b_bf16(vTB + (size_t)(n0 + t * 16) * DD, lane, DD);
#pragma unroll 2
    for (int kt = 0; kt < 16; ++kt) {
      const int kn = (kt + 1 < 16) ? (kt + 1) * 32 : kt * 32;
      v16bf a1 = load_a_f32(Sbuf + kn, lane, 512);
      v16bf b1[4];
#pragma unroll
      for (int t = 0; t < 4; ++t)
        b1[t] = load_b_bf16(vTB + (size_t)(n0 + t * 16) * DD + kn, lane, DD);
#pragma unroll
      for (int t = 0; t < 4; ++t)
        acc[t] = WMMA_BF16(a0, b0[t], acc[t]);
      a0 = a1;
#pragma unroll
      for (int t = 0; t < 4; ++t) b0[t] = b1[t];
    }
#pragma unroll
    for (int t = 0; t < 4; ++t)
#pragma unroll
      for (int i = 0; i < 8; ++i) {
        const int m = mb + i;
        const float val = acc[t][i] * inv_sum[m];
        outEN[((size_t)batch * LQ + row0 + m) * DD + n0 + t * 16 + n] = val;
        ebuf[m * 512 + n0 + t * 16 + n] = (__bf16)val;
      }
  }
  __syncthreads();

  // ---- phase 4: WC = EN . v (A = bf16 energy tile in LDS) ----
  {
    v8f acc[4];
#pragma unroll
    for (int t = 0; t < 4; ++t) acc[t] = {};
    v16bf a0 = load_a_bf16(ebuf, lane, DD);
    v16bf b0[4];
#pragma unroll
    for (int t = 0; t < 4; ++t)
      b0[t] = load_b_bf16(vTB + (size_t)(n0 + t * 16) * DD, lane, DD);
#pragma unroll 2
    for (int kt = 0; kt < 16; ++kt) {
      const int kn = (kt + 1 < 16) ? (kt + 1) * 32 : kt * 32;
      v16bf a1 = load_a_bf16(ebuf + kn, lane, DD);
      v16bf b1[4];
#pragma unroll
      for (int t = 0; t < 4; ++t)
        b1[t] = load_b_bf16(vTB + (size_t)(n0 + t * 16) * DD + kn, lane, DD);
#pragma unroll
      for (int t = 0; t < 4; ++t)
        acc[t] = WMMA_BF16(a0, b0[t], acc[t]);
      a0 = a1;
#pragma unroll
      for (int t = 0; t < 4; ++t) b0[t] = b1[t];
    }
#pragma unroll
    for (int t = 0; t < 4; ++t)
#pragma unroll
      for (int i = 0; i < 8; ++i)
        outWC[((size_t)batch * LQ + row0 + mb + i) * DD + n0 + t * 16 + n] =
            acc[t][i];
  }
}

// ---------------------------------------------------------------------------
// Host launcher
// ---------------------------------------------------------------------------
extern "C" void kernel_launch(void* const* d_in, const int* in_sizes, int n_in,
                              void* d_out, int out_size, void* d_ws, size_t ws_size,
                              hipStream_t stream) {
  (void)in_sizes; (void)n_in; (void)out_size; (void)ws_size;
  const float* pre = (const float*)d_in[0];
  const float* q   = (const float*)d_in[1];
  const float* v   = (const float*)d_in[2];
  float* out = (float*)d_out;
  // Output layout: [weightedContext | energy | precompute], each 32x1024x512.
  float* outWC = out;
  float* outEN = out + (size_t)PLANE;
  float* outPC = out + (size_t)2 * PLANE;

  // Workspace: bf16 copies (~134 MB): q row-major + transposed, v row-major
  // + transposed, pre row-major.
  __bf16* qb   = (__bf16*)d_ws;
  __bf16* qTb  = qb   + (size_t)NB * LQ * DD;
  __bf16* vb   = qTb  + (size_t)NB * DD * LQ;
  __bf16* vTb  = vb   + (size_t)NB * DD * DD;
  __bf16* preb = vTb  + (size_t)NB * DD * DD;

  cvt_tr_kernel<<<dim3(LQ / 32, DD / 32, NB), 256, 0, stream>>>(q, qb, qTb, LQ);
  cvt_tr_kernel<<<dim3(DD / 32, DD / 32, NB), 256, 0, stream>>>(v, vb, vTb, DD);
  cvt_kernel<<<PLANE / (256 * 4), 256, 0, stream>>>(pre, preb);

  const size_t smem1 = 65536 + 16384 + 64;          // 80.1 KB
  const size_t smem2 = 32768 + 2 * 16384 + 64;      // 64.1 KB
  hipFuncSetAttribute((const void*)attn_pre_kernel,
                      hipFuncAttributeMaxDynamicSharedMemorySize, (int)smem1);
  hipFuncSetAttribute((const void*)attn_ev_kernel,
                      hipFuncAttributeMaxDynamicSharedMemorySize, (int)smem2);

  attn_pre_kernel<<<dim3(LQ / 16, NB), 256, smem1, stream>>>(preb, qb, qTb, outPC);
  attn_ev_kernel<<<dim3(LQ / 16, NB), 256, smem2, stream>>>(qb, vb, vTb, outWC, outEN);
}
